// DualEncoder_60473139528555
// MI455X (gfx1250) — matrix operand order repaired and checked by reading, not compiled, not run on hardware
//
#include <hip/hip_runtime.h>

// Problem dims (from reference)
#define Bsz 64
#define Tt  160
#define Ee  512
#define Hh  1024
#define G4H 4096           // 4*H
#define KC  1536           // E + H (concat GEMM K)
#define NKT (KC / 32)      // 48 k-tiles of 32
#define KSTEP 2            // k-tiles per barrier stage (64 k values)
#define NGROUP (NKT / KSTEP)   // 24 groups; E = 8 groups, H = 16 groups (no straddle)
#define LROW 72            // u16 elems per LDS row: 64 data + 8 pad (144B stride, conflict-free)

typedef unsigned short u16;
typedef __attribute__((ext_vector_type(16))) __bf16 v16bf;
typedef __attribute__((ext_vector_type(8)))  float  v8f;
typedef __attribute__((ext_vector_type(8)))  u16    v8us;
typedef __attribute__((ext_vector_type(16))) u16    v16us;
typedef __attribute__((ext_vector_type(4)))  int    v4i;

union BfVec { v16us u; v16bf b; };

// ---- CDNA5 async global->LDS path (ASYNCcnt-tracked DMA, no VGPR round-trip)
#if defined(__has_builtin)
#  if __has_builtin(__builtin_amdgcn_global_load_async_to_lds_b128)
#    define HAS_ASYNC_LDS 1
#  endif
#endif
#ifndef HAS_ASYNC_LDS
#  define HAS_ASYNC_LDS 0
#endif

// Builtin signature (from hipcc diagnostic): arg0 = v4i in global AS (printed
// as "__device__"), arg1 = v4i in LDS AS, then imm offset + imm cpol.
typedef __attribute__((address_space(1))) v4i* gv4i_p;
typedef __attribute__((address_space(3))) v4i* lv4i_p;

__device__ __forceinline__ void stage16(const u16* __restrict__ src, u16* dst) {
#if HAS_ASYNC_LDS
    __builtin_amdgcn_global_load_async_to_lds_b128(
        (gv4i_p)(src), (lv4i_p)(dst), 0, 0);
    __builtin_amdgcn_global_load_async_to_lds_b128(
        (gv4i_p)(src + 8), (lv4i_p)(dst + 8), 0, 0);
#else
    v8us lo = *(const v8us*)src;
    v8us hi = *(const v8us*)(src + 8);
    *(v8us*)dst = lo;
    *(v8us*)(dst + 8) = hi;
#endif
}

__device__ __forceinline__ void wait_async0() {
#if HAS_ASYNC_LDS
#  if __has_builtin(__builtin_amdgcn_s_wait_asynccnt)
    __builtin_amdgcn_s_wait_asynccnt(0);
#  else
    asm volatile("s_wait_asynccnt 0x0" ::: "memory");
#  endif
#endif
}

__device__ __forceinline__ u16 f2bf(float f) {
    unsigned u = __float_as_uint(f);
    u += 0x7FFFu + ((u >> 16) & 1u);   // round-to-nearest-even
    return (u16)(u >> 16);
}
__device__ __forceinline__ float sigmoidf_(float x) { return 1.0f / (1.0f + __expf(-x)); }
__device__ __forceinline__ float tanhf_(float x)    { return 1.0f - 2.0f / (1.0f + __expf(2.0f * x)); }

// ---- Pack [Wih | Whh] row-major into bf16 "Wcat"[4096][1536] ----------------
__global__ void prep_wcat_kernel(const float* __restrict__ Wih,
                                 const float* __restrict__ Whh,
                                 u16* __restrict__ Wcat) {
    int idx = blockIdx.x * blockDim.x + threadIdx.x;      // < 4096*1536
    int k = idx % KC, r = idx / KC;
    float v = (k < Ee) ? Wih[r * Ee + k] : Whh[r * Hh + (k - Ee)];
    Wcat[idx] = f2bf(v);
}

// ---- Gather embeddings for both sequences into bf16 X[(s*T+t)*B + b][E] -----
__global__ void gather_kernel(const int* __restrict__ ctx,
                              const int* __restrict__ rsp,
                              const float* __restrict__ emb,
                              u16* __restrict__ X) {
    int idx = blockIdx.x * blockDim.x + threadIdx.x;      // < 2*T*B*E
    int e  = idx % Ee;  int r1 = idx / Ee;
    int b  = r1 % Bsz;  int r2 = r1 / Bsz;
    int t  = r2 % Tt;   int s  = r2 / Tt;
    int tok = (s == 0 ? ctx : rsp)[b * Tt + t];
    X[idx] = f2bf(emb[tok * Ee + e]);
}

__global__ void init_state_kernel(u16* __restrict__ Hbf0, u16* __restrict__ Hbf1,
                                  float* __restrict__ Hf, float* __restrict__ Cf) {
    int idx = blockIdx.x * blockDim.x + threadIdx.x;      // < 2*B*H
    Hbf0[idx] = 0; Hbf1[idx] = 0; Hf[idx] = 0.f; Cf[idx] = 0.f;
}

// ---- One LSTM timestep for both sequences (blockIdx.y = sequence) -----------
// gates[64,4096] = [x_t | h] @ Wcat^T. Each wave owns one 16(batch)x16(hidden)
// tile for ALL 4 gates. B tiles are DMA'd into LDS via async global->LDS loads
// (double-buffered, 2 k-tiles per barrier stage, s_wait_asynccnt placed after
// the WMMAs); A fragments are register double-buffered per group. The cell
// nonlinearity fuses in-register so gates never touch memory.
__global__ __launch_bounds__(256) void lstm_step_kernel(
    const u16* __restrict__ Xbf,   // [2*T*B*E] bf16
    const u16* __restrict__ Hprev, // [2*B*H] bf16 (read)
    u16*       __restrict__ Hnext, // [2*B*H] bf16 (write)
    float*     __restrict__ Hf,    // [2*B*H] f32
    float*     __restrict__ Cf,    // [2*B*H] f32
    const u16* __restrict__ Wcat,  // [4096*1536] bf16
    const float* __restrict__ bih, const float* __restrict__ bhh, int t)
{
    __shared__ __align__(32) u16 lds[2][128 * LROW];   // 2 x 18 KB

    const int tid   = threadIdx.x;
    const int s     = blockIdx.y;
    const int warp  = tid >> 5;
    const int lane  = tid & 31;
    const int lrow  = lane & 15;
    const int lhalf = lane >> 4;
    const int m0    = (warp & 3) * 16;                 // batch tile base
    const int jtl   = warp >> 2;                       // local j-tile (0/1)
    const int j0    = (blockIdx.x * 2 + jtl) * 16;     // hidden-unit tile base

    // ---- cooperative B-staging mapping: 256 threads DMA 16 KB per group -----
    const int c_chunk = tid & 1;           // 16-elem chunk within a 32-k sub-row
    const int c_row   = (tid >> 1) & 15;   // j within tile
    const int c_gate  = (tid >> 5) & 3;
    const int c_jtl   = tid >> 7;
    const u16* __restrict__ c_src =
        Wcat + (size_t)(c_gate * Hh + (blockIdx.x * 2 + c_jtl) * 16 + c_row) * KC
             + c_chunk * 16;
    const int c_dst = (((c_jtl * 4 + c_gate) * 16 + c_row) * LROW) + c_chunk * 16;

    // ---- A-fragment source pointers (x_t region then h region) --------------
    const int arow = m0 + lrow;
    const u16* __restrict__ axb = Xbf + (size_t)((s * Tt + t) * Bsz + arow) * Ee;  // + k0
    const u16* __restrict__ ahb = Hprev + (size_t)(s * Bsz + arow) * Hh - Ee;      // + k0

    v8f acc0 = {}, acc1 = {}, acc2 = {}, acc3 = {};
    v8us aLo[2][KSTEP], aHi[2][KSTEP];

    // Prologue: stage group 0 (from X region) into buf0, preload its A frags
    #pragma unroll
    for (int sub = 0; sub < KSTEP; ++sub) {
        stage16(c_src + sub * 32, &lds[0][c_dst + sub * 32]);
        aLo[0][sub] = *(const v8us*)(axb + sub * 32 + 16 * lhalf);
        aHi[0][sub] = *(const v8us*)(axb + sub * 32 + 16 * lhalf + 8);
    }
    wait_async0();
    __syncthreads();

    #pragma unroll 2
    for (int g = 0; g < NGROUP; ++g) {
        const int cur = g & 1;
        if (g + 1 < NGROUP) {                      // stage group g+1 (uniform branch)
            const int k1 = (g + 1) * (KSTEP * 32);
            const u16* src = c_src + k1;
            u16* dst = &lds[cur ^ 1][c_dst];
            const u16* abase = ((k1 < Ee) ? axb : ahb) + k1;
            #pragma unroll
            for (int sub = 0; sub < KSTEP; ++sub) {
                stage16(src + sub * 32, dst + sub * 32);
                aLo[cur ^ 1][sub] = *(const v8us*)(abase + sub * 32 + 16 * lhalf);
                aHi[cur ^ 1][sub] = *(const v8us*)(abase + sub * 32 + 16 * lhalf + 8);
            }
        }

        const int rbase = (jtl * 4) * 16 + lrow;
        #pragma unroll
        for (int sub = 0; sub < KSTEP; ++sub) {
            BfVec a;
            #pragma unroll
            for (int i = 0; i < 8; ++i) { a.u[i] = aLo[cur][sub][i]; a.u[i + 8] = aHi[cur][sub][i]; }

            const int soff = sub * 32 + 16 * lhalf;
            BfVec b0, b1, b2, b3;
            {
                v8us lo, hi; int off;
                off = (rbase + 0 * 16) * LROW + soff;
                lo = *(const v8us*)&lds[cur][off]; hi = *(const v8us*)&lds[cur][off + 8];
                #pragma unroll
                for (int i = 0; i < 8; ++i) { b0.u[i] = lo[i]; b0.u[i + 8] = hi[i]; }
                off = (rbase + 1 * 16) * LROW + soff;
                lo = *(const v8us*)&lds[cur][off]; hi = *(const v8us*)&lds[cur][off + 8];
                #pragma unroll
                for (int i = 0; i < 8; ++i) { b1.u[i] = lo[i]; b1.u[i + 8] = hi[i]; }
                off = (rbase + 2 * 16) * LROW + soff;
                lo = *(const v8us*)&lds[cur][off]; hi = *(const v8us*)&lds[cur][off + 8];
                #pragma unroll
                for (int i = 0; i < 8; ++i) { b2.u[i] = lo[i]; b2.u[i + 8] = hi[i]; }
                off = (rbase + 3 * 16) * LROW + soff;
                lo = *(const v8us*)&lds[cur][off]; hi = *(const v8us*)&lds[cur][off + 8];
                #pragma unroll
                for (int i = 0; i < 8; ++i) { b3.u[i] = lo[i]; b3.u[i + 8] = hi[i]; }
            }

            acc0 = __builtin_amdgcn_wmma_f32_16x16x32_bf16(false, a.b, false, b0.b, (short)0, acc0, false, false);
            acc1 = __builtin_amdgcn_wmma_f32_16x16x32_bf16(false, a.b, false, b1.b, (short)0, acc1, false, false);
            acc2 = __builtin_amdgcn_wmma_f32_16x16x32_bf16(false, a.b, false, b2.b, (short)0, acc2, false, false);
            acc3 = __builtin_amdgcn_wmma_f32_16x16x32_bf16(false, a.b, false, b3.b, (short)0, acc3, false, false);
        }

        if (g + 1 < NGROUP) {
            wait_async0();       // staged data for g+1 is in LDS (after the WMMAs)
            __syncthreads();     // buf[cur] consumed by all; buf[cur^1] visible to all
        }
    }

    // Fused LSTM cell update. Acc layout: VGPR r, lane l -> row M = r + 8*(l/16),
    // col N = l%16. Each (batch b, hidden j) has a unique owner lane/vgpr.
    const int j = j0 + lrow;
    const float bi = bih[j]          + bhh[j];
    const float bf = bih[Hh + j]     + bhh[Hh + j];
    const float bg = bih[2 * Hh + j] + bhh[2 * Hh + j];
    const float bo = bih[3 * Hh + j] + bhh[3 * Hh + j];

    #pragma unroll
    for (int r = 0; r < 8; ++r) {
        const int bidx = m0 + r + 8 * lhalf;
        const int off  = (s * Bsz + bidx) * Hh + j;
        const float ig = sigmoidf_(acc0[r] + bi);
        const float fg = sigmoidf_(acc1[r] + bf);
        const float gg = tanhf_(acc2[r] + bg);
        const float og = sigmoidf_(acc3[r] + bo);
        const float cn = fg * Cf[off] + ig * gg;
        const float hn = og * tanhf_(cn);
        Cf[off] = cn;
        Hf[off] = hn;
        Hnext[off] = f2bf(hn);
    }
}

// ---- out[b] = sigmoid( ch[b] . (M . rh[b]) ) --------------------------------
__global__ __launch_bounds__(256) void bilinear_kernel(const float* __restrict__ Hf,
                                                       const float* __restrict__ M,
                                                       float* __restrict__ out) {
    __shared__ float rsh[Hh];
    __shared__ float red[256];
    const int b = blockIdx.x, tid = threadIdx.x;
    for (int jj = tid; jj < Hh; jj += 256) rsh[jj] = Hf[(Bsz + b) * Hh + jj];  // rh (seq 1)
    __syncthreads();
    float accum = 0.f;
    for (int i = tid; i < Hh; i += 256) {
        const float4* mr = (const float4*)(M + (size_t)i * Hh);
        const float4* rr = (const float4*)rsh;
        float dot = 0.f;
        #pragma unroll 4
        for (int j4 = 0; j4 < Hh / 4; ++j4) {
            float4 m4 = mr[j4]; float4 r4 = rr[j4];
            dot += m4.x * r4.x + m4.y * r4.y + m4.z * r4.z + m4.w * r4.w;
        }
        accum += Hf[b * Hh + i] * dot;  // ch (seq 0)
    }
    red[tid] = accum;
    __syncthreads();
    for (int off = 128; off > 0; off >>= 1) {
        if (tid < off) red[tid] += red[tid + off];
        __syncthreads();
    }
    if (tid == 0) out[b] = sigmoidf_(red[0]);
}

extern "C" void kernel_launch(void* const* d_in, const int* in_sizes, int n_in,
                              void* d_out, int out_size, void* d_ws, size_t ws_size,
                              hipStream_t stream) {
    (void)in_sizes; (void)n_in; (void)out_size; (void)ws_size;
    const int*   ctx = (const int*)  d_in[0];
    const int*   rsp = (const int*)  d_in[1];
    const float* emb = (const float*)d_in[2];
    const float* Wih = (const float*)d_in[3];
    const float* Whh = (const float*)d_in[4];
    const float* bih = (const float*)d_in[5];
    const float* bhh = (const float*)d_in[6];
    const float* M   = (const float*)d_in[7];
    float* out = (float*)d_out;

    char* p = (char*)d_ws;
    auto alloc = [&](size_t bytes) -> char* {
        char* q = p; p += (bytes + 255) & ~(size_t)255; return q;
    };
    u16*   Wcat = (u16*)  alloc((size_t)G4H * KC * 2);            // 12 MB
    u16*   Xbf  = (u16*)  alloc((size_t)2 * Tt * Bsz * Ee * 2);   // 21 MB
    u16*   Hbf0 = (u16*)  alloc((size_t)2 * Bsz * Hh * 2);
    u16*   Hbf1 = (u16*)  alloc((size_t)2 * Bsz * Hh * 2);
    float* Hf   = (float*)alloc((size_t)2 * Bsz * Hh * 4);
    float* Cf   = (float*)alloc((size_t)2 * Bsz * Hh * 4);

    prep_wcat_kernel<<<(G4H * KC) / 256, 256, 0, stream>>>(Wih, Whh, Wcat);
    gather_kernel<<<(2 * Tt * Bsz * Ee) / 256, 256, 0, stream>>>(ctx, rsp, emb, Xbf);
    init_state_kernel<<<(2 * Bsz * Hh) / 256, 256, 0, stream>>>(Hbf0, Hbf1, Hf, Cf);

    for (int t = 0; t < Tt; ++t) {
        u16* hprev = (t & 1) ? Hbf1 : Hbf0;
        u16* hnext = (t & 1) ? Hbf0 : Hbf1;
        // gridDim.y = 2: both sequences advance concurrently
        lstm_step_kernel<<<dim3(32, 2), 256, 0, stream>>>(Xbf, hprev, hnext, Hf, Cf,
                                                          Wcat, bih, bhh, t);
    }
    bilinear_kernel<<<Bsz, 256, 0, stream>>>(Hf, M, out);
}
